// MoEDQN_51170240365280
// MI455X (gfx1250) — matrix operand order repaired and checked by reading, not compiled, not run on hardware
//
#include <hip/hip_runtime.h>
#include <math.h>

typedef __attribute__((ext_vector_type(16))) _Float16 v16h;
typedef __attribute__((ext_vector_type(8)))  float    v8f;

#if defined(__has_builtin)
#if __has_builtin(__builtin_amdgcn_tensor_load_to_lds) && \
    __has_builtin(__builtin_amdgcn_s_wait_tensorcnt)
#define USE_TDM 1
#endif
#endif

// ---------------------------------------------------------------------------
// fp32 -> fp16 bulk convert (4 elements / thread, vectorized)
// ---------------------------------------------------------------------------
__global__ void cvt_f32_f16_kernel(const float* __restrict__ in,
                                   _Float16* __restrict__ out, long long n4)
{
    const long long i = (long long)blockIdx.x * blockDim.x + threadIdx.x;
    if (i >= n4) return;
    const float4 v = ((const float4*)in)[i];
    union { _Float16 h[4]; int2 q; } u;
    u.h[0] = (_Float16)v.x; u.h[1] = (_Float16)v.y;
    u.h[2] = (_Float16)v.z; u.h[3] = (_Float16)v.w;
    ((int2*)out)[i] = u.q;
}

// fp32 (bz,K,N) -> fp16 (bz,N,K): puts einsum weights into (N,K) row-major
__global__ void cvt_transpose_kernel(const float* __restrict__ in,
                                     _Float16* __restrict__ out,
                                     int K, int N, long long total)
{
    const long long i = (long long)blockIdx.x * blockDim.x + threadIdx.x;
    if (i >= total) return;
    const long long nk = (long long)N * K;
    const long long bz = i / nk;
    const long long rem = i - bz * nk;
    const int n = (int)(rem / K);
    const int k = (int)(rem - (long long)n * K);
    out[i] = (_Float16)in[(bz * K + k) * (long long)N + n];
}

// ---------------------------------------------------------------------------
// WMMA fragment helpers (wave32 layouts, CDNA5 ISA §7.12.2).
// ---------------------------------------------------------------------------
__device__ inline v16h a_frag(const _Float16* As, int waveRow0, int lane)
{
    union { int4 q[2]; v16h v; } u;
    const _Float16* p = &As[(waveRow0 + (lane & 15)) * 32 + ((lane >> 4) << 3)];
    u.q[0] = *(const int4*)p;
    u.q[1] = *(const int4*)(p + 16);
    return u.v;
}
__device__ inline v16h b_frag(const _Float16* Bs, int sub, int lane)
{
    union { int4 q[2]; v16h v; } u;
    const _Float16* p = &Bs[((sub << 4) + (lane & 15)) * 32 + ((lane >> 4) << 3)];
    u.q[0] = *(const int4*)p;
    u.q[1] = *(const int4*)(p + 16);
    return u.v;
}

#ifdef USE_TDM
// ---------------------------------------------------------------------------
// Issue one TDM 2D tile load: global (rows x cols, row stride ldElems, f16)
// -> LDS at byte offset ldsAddr, tile = tile1 rows x tile0 elems, row-major.
// OOB rows/cols (beyond dim0Rem/rowsRem) are zero-filled by the TDM.
// D# layout per CDNA5 ISA §8.3/§8.4 (group0 128b, group1 256b).
// Builtin arity differs by toolchain: clang<=22 (ROCm 7.2) = 5 args,
// clang>=23 (amdgpu-toolchain/therock) = 6 args (extra int32x8 before cpol).
// ---------------------------------------------------------------------------
typedef unsigned int u32x4 __attribute__((ext_vector_type(4)));
typedef int          i32x8 __attribute__((ext_vector_type(8)));
typedef int          i32x4 __attribute__((ext_vector_type(4)));

__device__ inline void tdm_load_2d(unsigned ldsAddr, const _Float16* gptr,
                                   unsigned dim0Rem, unsigned ldElems,
                                   unsigned rowsRem, unsigned tile0, unsigned tile1)
{
    const unsigned long long ga = (unsigned long long)gptr;
    u32x4 g0;
    g0[0] = 1u;                                      // count=1, user descriptor
    g0[1] = ldsAddr;                                 // lds_addr (bytes)
    g0[2] = (unsigned)(ga & 0xFFFFFFFFu);            // global_addr[31:0]
    g0[3] = (unsigned)((ga >> 32) & 0x01FFFFFFu)     // global_addr[56:32]
          | (2u << 30);                              // type = 2 ("image")
    i32x8 g1;
    g1[0] = 0x00010000;                              // data_size=1 (2 bytes)
    g1[1] = (int)((dim0Rem & 0xFFFFu) << 16);        // tensor_dim0[15:0]
    g1[2] = (int)((dim0Rem >> 16) | ((rowsRem & 0xFFFFu) << 16)); // dim0 hi | dim1 lo
    g1[3] = (int)((rowsRem >> 16) | (tile0 << 16));  // dim1 hi | tile_dim0
    g1[4] = (int)tile1;                              // tile_dim1 (tile_dim2 = 0)
    g1[5] = (int)ldElems;                            // tensor_dim0_stride[31:0]
    g1[6] = 0;                                       // stride hi | dim1_stride lo
    g1[7] = 0;
    const i32x4 z4 = {0, 0, 0, 0};
#if __clang_major__ >= 23
    const i32x8 z8 = {0, 0, 0, 0, 0, 0, 0, 0};
    __builtin_amdgcn_tensor_load_to_lds(g0, g1, z4, z4, z8, 0);
#else
    __builtin_amdgcn_tensor_load_to_lds(g0, g1, z4, z4, 0);
#endif
}
#endif

// ---------------------------------------------------------------------------
// Generic WMMA GEMM, all-f16 operands, f32 accumulate.
//   C[m,n] = act( sum_k A[m,k]*B[n,k] + bias[n] )   (B is (N,K) row-major f16)
//   Block: 256 thr = 8 waves; tile 128(M) x 32(N); two accumulators per wave.
//   Staging: TDM double-buffered tensor_load_to_lds (wave 0 drives the DMA),
//   synchronized with s_wait_tensorcnt + workgroup barrier.
//   NOTE: requires lda/ldb to equal the packed row length (true for all calls).
// ---------------------------------------------------------------------------
__global__ void __launch_bounds__(256)
gemm_wmma(const _Float16* __restrict__ A, int lda, long long aBatch,
          const _Float16* __restrict__ Bw, int ldb, long long bBatch,
          const float* __restrict__ bias, long long biasBatch,
          float* __restrict__ C, _Float16* __restrict__ C16,
          int ldc, long long cBatch,
          int M, int N, int K, int relu)
{
    __shared__ _Float16 As[2][128 * 32];   // 16 KB ping-pong
    __shared__ _Float16 Bs[2][32 * 32];    //  4 KB ping-pong

    const int t = threadIdx.x, w = t >> 5, lane = t & 31;
    const long long aOff = (long long)blockIdx.z * aBatch;
    const long long bOff = (long long)blockIdx.z * bBatch;
    const long long cOff = (long long)blockIdx.z * cBatch;
    const long long biasOff = (long long)blockIdx.z * biasBatch;
    const int mBlock = blockIdx.x * 128;
    const int n0     = blockIdx.y * 32;

    const _Float16* aBase = A + aOff + (long long)mBlock * lda;
    const _Float16* bBase = Bw + bOff + (long long)n0 * ldb;
    const unsigned aRows = (unsigned)(M - mBlock);
    const unsigned bRows = (unsigned)((N > n0) ? (N - n0) : 0);

    v8f acc[2];
    acc[0] = (v8f){0.f,0.f,0.f,0.f,0.f,0.f,0.f,0.f};
    acc[1] = acc[0];

#ifdef USE_TDM
    const unsigned asAddr = (unsigned)(unsigned long long)(const void*)&As[0][0];
    const unsigned bsAddr = (unsigned)(unsigned long long)(const void*)&Bs[0][0];
    // prologue: DMA tile k0=0 into buffer 0 (wave 0 drives the TDM)
    if (w == 0) {
        tdm_load_2d(asAddr, aBase, (unsigned)K, (unsigned)lda, aRows, 32u, 128u);
        tdm_load_2d(bsAddr, bBase, (unsigned)K, (unsigned)ldb, bRows, 32u, 32u);
    }
    int buf = 0;
    for (int k0 = 0; k0 < K; k0 += 32) {
        if (w == 0) {
            const int nk = k0 + 32;
            if (nk < K) {
                const int nb = buf ^ 1;
                tdm_load_2d(asAddr + (unsigned)nb * 128u * 32u * 2u, aBase + nk,
                            (unsigned)(K - nk), (unsigned)lda, aRows, 32u, 128u);
                tdm_load_2d(bsAddr + (unsigned)nb * 32u * 32u * 2u, bBase + nk,
                            (unsigned)(K - nk), (unsigned)ldb, bRows, 32u, 32u);
                __builtin_amdgcn_s_wait_tensorcnt(2);   // current tile complete
            } else {
                __builtin_amdgcn_s_wait_tensorcnt(0);   // last tile complete
            }
        }
        __syncthreads();

        const v16h af  = a_frag(As[buf], w * 16, lane);
        const v16h bf0 = b_frag(Bs[buf], 0, lane);
        const v16h bf1 = b_frag(Bs[buf], 1, lane);
        acc[0] = __builtin_amdgcn_wmma_f32_16x16x32_f16(false, af, false, bf0,
                                                        (short)0, acc[0], false, false);
        acc[1] = __builtin_amdgcn_wmma_f32_16x16x32_f16(false, af, false, bf1,
                                                        (short)0, acc[1], false, false);
        __syncthreads();
        buf ^= 1;
    }
#else
    // fallback: cooperative b128 staging through VGPRs
    const int arow = t >> 1, ajb = (t & 1) << 4;
    const bool avalid = (mBlock + arow) < M;
    const _Float16* aRow = aBase + (long long)arow * lda;
    const int brow = t >> 2, bjb = (t & 3) << 3;
    const bool bvalid = (t < 128) && ((n0 + brow) < N);
    const _Float16* bRow = bBase + (long long)brow * ldb;
    const int4 z4 = {0, 0, 0, 0};
    for (int k0 = 0; k0 < K; k0 += 32) {
        int4 a0 = z4, a1 = z4;
        if (avalid) {
            const int4* p = (const int4*)(aRow + k0 + ajb);
            a0 = p[0]; a1 = p[1];
        }
        *(int4*)&As[0][arow * 32 + ajb]     = a0;
        *(int4*)&As[0][arow * 32 + ajb + 8] = a1;
        if (t < 128) {
            const int4 b0 = bvalid ? *(const int4*)(bRow + k0 + bjb) : z4;
            *(int4*)&Bs[0][brow * 32 + bjb] = b0;
        }
        __syncthreads();
        const v16h af  = a_frag(As[0], w * 16, lane);
        const v16h bf0 = b_frag(Bs[0], 0, lane);
        const v16h bf1 = b_frag(Bs[0], 1, lane);
        acc[0] = __builtin_amdgcn_wmma_f32_16x16x32_f16(false, af, false, bf0,
                                                        (short)0, acc[0], false, false);
        acc[1] = __builtin_amdgcn_wmma_f32_16x16x32_f16(false, af, false, bf1,
                                                        (short)0, acc[1], false, false);
        __syncthreads();
    }
#endif

    // C/D layout: lane = N column, VGPR r = M row (lanes>=16 -> rows 8..15)
    const int mbase = mBlock + w * 16 + ((lane >> 4) ? 8 : 0);
#pragma unroll
    for (int s = 0; s < 2; ++s) {
        const int nG = n0 + (s << 4) + (lane & 15);
        if (nG < N) {
            const float bv = bias ? bias[biasOff + nG] : 0.f;
#pragma unroll
            for (int r = 0; r < 8; ++r) {
                const int m = mbase + r;
                if (m < M) {
                    float v = acc[s][r] + bv;
                    if (relu) v = fmaxf(v, 0.f);
                    const long long ci = cOff + (long long)m * ldc + nG;
                    if (C)   C[ci]   = v;
                    if (C16) C16[ci] = (_Float16)v;
                }
            }
        }
    }
}

// ---------------------------------------------------------------------------
// Implicit-GEMM conv (VALID, NCHW, f16 in/out), fused bias + ReLU.
// im2col tap offsets precomputed once into LDS (K <= 576).
// ---------------------------------------------------------------------------
__global__ void __launch_bounds__(256)
conv_wmma(const _Float16* __restrict__ X, const _Float16* __restrict__ Wgt,
          const float* __restrict__ bias, _Float16* __restrict__ Y,
          int Cin, int H, int Wd, int Cout, int OH, int OW,
          int KH, int KW, int stride, int M, int K)
{
    __shared__ _Float16 As[128 * 32];
    __shared__ _Float16 Bs[32 * 32];
    __shared__ int offT[576];

    const int t = threadIdx.x, w = t >> 5, lane = t & 31;
    const int mBlock = blockIdx.x * 128;
    const int n0     = blockIdx.y * 32;
    const int pixels = OH * OW;
    const int kwin   = KH * KW;

    for (int k = t; k < K; k += 256) {
        const int c  = k / kwin;
        const int rs = k - c * kwin;
        const int r  = rs / KW, s = rs - r * KW;
        offT[k] = (c * H + r) * Wd + s;
    }

    const int arow = t >> 1, ajb = (t & 1) << 4;
    const int am = mBlock + arow;
    const bool avalid = am < M;
    const int ab  = am / pixels;
    const int ap  = am - ab * pixels;
    const int aoy = ap / OW, aox = ap - aoy * OW;
    const _Float16* xbase =
        X + ((long long)ab * Cin * H + aoy * stride) * Wd + aox * stride;
    const int brow = t >> 2, bjb = (t & 3) << 3;
    const int bnG = n0 + brow;
    const bool bvalid = (t < 128) && (bnG < Cout);
    const _Float16* bRow = Wgt + (long long)bnG * K;

    v8f acc[2];
    acc[0] = (v8f){0.f,0.f,0.f,0.f,0.f,0.f,0.f,0.f};
    acc[1] = acc[0];
    const int4 z4 = {0, 0, 0, 0};

    __syncthreads();   // offT ready

    for (int k0 = 0; k0 < K; k0 += 32) {
        __align__(16) _Float16 tmp[16];
#pragma unroll
        for (int j = 0; j < 16; ++j) {
            const _Float16 zh = (_Float16)0.f;
            tmp[j] = avalid ? xbase[offT[k0 + ajb + j]] : zh;
        }
        *(int4*)&As[arow * 32 + ajb]     = *(const int4*)&tmp[0];
        *(int4*)&As[arow * 32 + ajb + 8] = *(const int4*)&tmp[8];
        if (t < 128) {
            const int4 b0 = bvalid ? *(const int4*)(bRow + k0 + bjb) : z4;
            *(int4*)&Bs[brow * 32 + bjb] = b0;
        }
        __syncthreads();

        const v16h af  = a_frag(As, w * 16, lane);
        const v16h bf0 = b_frag(Bs, 0, lane);
        const v16h bf1 = b_frag(Bs, 1, lane);
        acc[0] = __builtin_amdgcn_wmma_f32_16x16x32_f16(false, af, false, bf0,
                                                        (short)0, acc[0], false, false);
        acc[1] = __builtin_amdgcn_wmma_f32_16x16x32_f16(false, af, false, bf1,
                                                        (short)0, acc[1], false, false);
        __syncthreads();
    }

    const int mbase = mBlock + w * 16 + ((lane >> 4) ? 8 : 0);
#pragma unroll
    for (int s = 0; s < 2; ++s) {
        const int nG = n0 + (s << 4) + (lane & 15);
        if (nG < Cout) {
            const float bv = bias[nG];
#pragma unroll
            for (int r = 0; r < 8; ++r) {
                const int m = mbase + r;
                if (m < M) {
                    const int b  = m / pixels;
                    const int p  = m - b * pixels;
                    const int oy = p / OW, ox = p - oy * OW;
                    const float v = fmaxf(acc[s][r] + bv, 0.f);
                    Y[(((long long)b * Cout + nG) * OH + oy) * OW + ox] = (_Float16)v;
                }
            }
        }
    }
}

// ---------------------------------------------------------------------------
// GRU cell elementwise; writes f32 new_hidden (d_out) + f16 copy (ws).
// ---------------------------------------------------------------------------
__global__ void gru_kernel(const float* __restrict__ gi, const float* __restrict__ gh,
                           const float* __restrict__ h_in, float* __restrict__ h_out,
                           _Float16* __restrict__ h16, int total)
{
    const int i = blockIdx.x * blockDim.x + threadIdx.x;
    if (i >= total) return;
    const int b = i >> 7;       // RH = 128
    const int j = i & 127;
    const long long gb = (long long)b * 384;
    const float ir = gi[gb + j],       hr = gh[gb + j];
    const float iz = gi[gb + 128 + j], hz = gh[gb + 128 + j];
    const float in = gi[gb + 256 + j], hn = gh[gb + 256 + j];
    const float r = 1.f / (1.f + expf(-(ir + hr)));
    const float z = 1.f / (1.f + expf(-(iz + hz)));
    const float n = tanhf(in + r * hn);
    const float h = (1.f - z) * n + z * h_in[(long long)b * 128 + j];
    h_out[(long long)b * 128 + j] = h;
    h16[(long long)b * 128 + j]   = (_Float16)h;
}

// ---------------------------------------------------------------------------
// Softmax over 8 logits, top-k (k from device scalar), sparse gate weights.
// ---------------------------------------------------------------------------
__global__ void gate_kernel(const float* __restrict__ logits, const int* __restrict__ topk,
                            float* __restrict__ gates_out, float* __restrict__ idx_out,
                            float* __restrict__ wts, int Bn)
{
    const int b = blockIdx.x * blockDim.x + threadIdx.x;
    if (b >= Bn) return;
    float p[8], tmp[8];
    float mx = -1e30f;
#pragma unroll
    for (int e = 0; e < 8; ++e) { p[e] = logits[b * 8 + e]; mx = fmaxf(mx, p[e]); }
    float s = 0.f;
#pragma unroll
    for (int e = 0; e < 8; ++e) { p[e] = expf(p[e] - mx); s += p[e]; }
    const float inv = 1.f / s;
#pragma unroll
    for (int e = 0; e < 8; ++e) {
        p[e] *= inv;
        tmp[e] = p[e];
        gates_out[b * 8 + e] = p[e];
        wts[b * 8 + e] = 0.f;
    }
    int K = topk ? *topk : 2;
    K = (K < 0) ? 0 : (K > 8 ? 8 : K);
    const int iters = (K > 2) ? K : 2;
    for (int k = 0; k < iters; ++k) {
        int   aj = 0;
        float av = tmp[0];
#pragma unroll
        for (int e = 1; e < 8; ++e) if (tmp[e] > av) { av = tmp[e]; aj = e; }
        if (k < K) wts[b * 8 + aj] = p[aj];
        if (k < 2) idx_out[b * 2 + k] = (float)aj;
        tmp[aj] = -1.f;
    }
}

// ---------------------------------------------------------------------------
// Sparse combine: q[b,a] = sum_e wts[b,e] * q_all[e,b,a]
// ---------------------------------------------------------------------------
__global__ void combine_kernel(const float* __restrict__ wts, const float* __restrict__ qall,
                               float* __restrict__ q, int Bn)
{
    const int i = blockIdx.x * blockDim.x + threadIdx.x;
    if (i >= Bn * 18) return;
    const int b = i / 18, a = i - b * 18;
    float s = 0.f;
#pragma unroll
    for (int e = 0; e < 8; ++e)
        s += wts[b * 8 + e] * qall[((long long)e * Bn + b) * 18 + a];
    q[i] = s;
}

// ---------------------------------------------------------------------------
extern "C" void kernel_launch(void* const* d_in, const int* in_sizes, int n_in,
                              void* d_out, int out_size, void* d_ws, size_t ws_size,
                              hipStream_t stream)
{
    const float* obs    = (const float*)d_in[0];
    const float* hidden = (const float*)d_in[1];
    const int*   topk   = (const int*)d_in[2];
    const float* cw1 = (const float*)d_in[3];  const float* cb1 = (const float*)d_in[4];
    const float* cw2 = (const float*)d_in[5];  const float* cb2 = (const float*)d_in[6];
    const float* cw3 = (const float*)d_in[7];  const float* cb3 = (const float*)d_in[8];
    const float* fcw = (const float*)d_in[9];  const float* fcb = (const float*)d_in[10];
    const float* Wi  = (const float*)d_in[11]; const float* Wh  = (const float*)d_in[12];
    const float* bi  = (const float*)d_in[13]; const float* bh  = (const float*)d_in[14];
    const float* Wo  = (const float*)d_in[15]; const float* bo  = (const float*)d_in[16];
    const float* We1 = (const float*)d_in[17]; const float* be1 = (const float*)d_in[18];
    const float* We2 = (const float*)d_in[19]; const float* be2 = (const float*)d_in[20];
    (void)in_sizes; (void)n_in; (void)out_size; (void)ws_size;

    char* wsb  = (char*)d_ws;
    float* out = (float*)d_out;
    const size_t MB = 1024ull * 1024ull;

    _Float16* obs_h  = (_Float16*)(wsb + 0);             // 115.6 MB (dead after conv1)
    _Float16* c1_h   = (_Float16*)(wsb + 120 * MB);      // 52.4 MB
    _Float16* c2_h   = (_Float16*)(wsb + 0);             // 21.2 MB (over obs_h)
    _Float16* c3_h   = (_Float16*)(wsb + 120 * MB);      // 12.8 MB (over c1_h)
    _Float16* feats_h= (_Float16*)(wsb + 0);             // 2.1 MB  (over c2_h)
    float*    gi     = (float*)   (wsb + 4 * MB);
    float*    gh     = (float*)   (wsb + 8 * MB);
    _Float16* hid_h  = (_Float16*)(wsb + 12 * MB);
    float*    logits = (float*)   (wsb + 13 * MB);
    float*    wts    = (float*)   (wsb + 14 * MB);
    _Float16* h1_h   = (_Float16*)(wsb + 15 * MB);       // 8.4 MB
    float*    qall   = (float*)   (wsb + 24 * MB);       // 1.2 MB
    const size_t W0  = 173 * MB;
    _Float16* cw1_h  = (_Float16*)(wsb + W0);
    _Float16* cw2_h  = (_Float16*)(wsb + W0 + 64  * 1024);
    _Float16* cw3_h  = (_Float16*)(wsb + W0 + 128 * 1024);
    _Float16* fcw_h  = (_Float16*)(wsb + W0 + 256 * 1024);
    _Float16* Wi_h   = (_Float16*)(wsb + W0 + 4 * MB);
    _Float16* Wh_h   = (_Float16*)(wsb + W0 + 4 * MB + 512 * 1024);
    _Float16* Wo_h   = (_Float16*)(wsb + W0 + 4 * MB + 768 * 1024);
    _Float16* We1t_h = (_Float16*)(wsb + W0 + 5 * MB);
    _Float16* We2t_h = (_Float16*)(wsb + W0 + 7 * MB + 256 * 1024);
    _Float16* hidden_h=(_Float16*)(wsb + W0 + 7 * MB + 512 * 1024);

    float* q_out    = out;
    float* gate_out = out + 36864;
    float* hid_out  = out + 53248;
    float* idx_out  = out + 315392;

    const dim3 blk(256);

    // one-shot fp32 -> fp16 conversions
    cvt_f32_f16_kernel<<<56448, blk, 0, stream>>>(obs, obs_h, 14450688LL);
    cvt_f32_f16_kernel<<<8,    blk, 0, stream>>>(cw1, cw1_h, 2048LL);
    cvt_f32_f16_kernel<<<32,   blk, 0, stream>>>(cw2, cw2_h, 8192LL);
    cvt_f32_f16_kernel<<<36,   blk, 0, stream>>>(cw3, cw3_h, 9216LL);
    cvt_f32_f16_kernel<<<1568, blk, 0, stream>>>(fcw, fcw_h, 401408LL);
    cvt_f32_f16_kernel<<<192,  blk, 0, stream>>>(Wi,  Wi_h,  49152LL);
    cvt_f32_f16_kernel<<<48,   blk, 0, stream>>>(Wh,  Wh_h,  12288LL);
    cvt_f32_f16_kernel<<<1,    blk, 0, stream>>>(Wo,  Wo_h,  256LL);
    cvt_f32_f16_kernel<<<256,  blk, 0, stream>>>(hidden, hidden_h, 65536LL);
    cvt_transpose_kernel<<<4096, blk, 0, stream>>>(We1, We1t_h, 512, 256, 1048576LL);
    cvt_transpose_kernel<<<144,  blk, 0, stream>>>(We2, We2t_h, 256, 18,  36864LL);

    // ConvEncoder (implicit-GEMM WMMA, f16)
    conv_wmma<<<dim3(6400, 1, 1), blk, 0, stream>>>(obs_h, cw1_h, cb1, c1_h,
        4, 84, 84, 32, 20, 20, 8, 8, 4, 819200, 256);
    conv_wmma<<<dim3(1296, 2, 1), blk, 0, stream>>>(c1_h, cw2_h, cb2, c2_h,
        32, 20, 20, 64, 9, 9, 4, 4, 2, 165888, 512);
    conv_wmma<<<dim3(784, 2, 1), blk, 0, stream>>>(c2_h, cw3_h, cb3, c3_h,
        64, 9, 9, 64, 7, 7, 3, 3, 1, 100352, 576);

    // FC (conv3 NCHW flatten is (B,3136) row-major), f16 out
    gemm_wmma<<<dim3(16, 16, 1), blk, 0, stream>>>(c3_h, 3136, 0, fcw_h, 3136, 0,
        fcb, 0, (float*)nullptr, feats_h, 512, 0, 2048, 512, 3136, 1);

    // GRU gate GEMMs (f32 out for elementwise cell)
    gemm_wmma<<<dim3(16, 12, 1), blk, 0, stream>>>(feats_h, 512, 0, Wi_h, 512, 0,
        bi, 0, gi, (_Float16*)nullptr, 384, 0, 2048, 384, 512, 0);
    gemm_wmma<<<dim3(16, 12, 1), blk, 0, stream>>>(hidden_h, 128, 0, Wh_h, 128, 0,
        bh, 0, gh, (_Float16*)nullptr, 384, 0, 2048, 384, 128, 0);
    gru_kernel<<<1024, blk, 0, stream>>>(gi, gh, hidden, hid_out, hid_h, 262144);

    // Router logits + gating
    gemm_wmma<<<dim3(16, 1, 1), blk, 0, stream>>>(hid_h, 128, 0, Wo_h, 128, 0,
        bo, 0, logits, (_Float16*)nullptr, 8, 0, 2048, 8, 128, 0);
    gate_kernel<<<8, blk, 0, stream>>>(logits, topk, gate_out, idx_out, wts, 2048);

    // Expert MLPs, batched over E via blockIdx.z
    gemm_wmma<<<dim3(16, 8, 8), blk, 0, stream>>>(feats_h, 512, 0,
        We1t_h, 512, 256LL * 512, be1, 256,
        (float*)nullptr, h1_h, 256, 2048LL * 256, 2048, 256, 512, 1);
    gemm_wmma<<<dim3(16, 1, 8), blk, 0, stream>>>(h1_h, 256, 2048LL * 256,
        We2t_h, 256, 18LL * 256, be2, 18,
        qall, (_Float16*)nullptr, 18, 2048LL * 18, 2048, 18, 256, 0);

    // Sparse combine
    combine_kernel<<<144, blk, 0, stream>>>(wts, qall, q_out, 2048);
}
